// GlobalAttentionModule_65592740544977
// MI455X (gfx1250) — matrix operand order repaired and checked by reading, not compiled, not run on hardware
//
#include <hip/hip_runtime.h>
#include <math.h>

// ---------------------------------------------------------------------------
// CDNA5 / gfx1250: bf16 WMMA implicit-GEMM conv pipeline + projection GEMM
// ---------------------------------------------------------------------------

typedef __attribute__((ext_vector_type(16))) __bf16 v16bf;
typedef __attribute__((ext_vector_type(8)))  __bf16 v8bf;
typedef __attribute__((ext_vector_type(8)))  float  v8f;

#define EPSV 1e-5f

__device__ __forceinline__ float gelu_f(float x) {
  return 0.5f * x * (1.0f + erff(x * 0.70710678118654752f));
}

// Assemble a v16bf WMMA fragment from two contiguous 8-element chunks.
// Mapping follows the 16-bit 16x32 A layout (and the mirrored B layout):
//   elems 0..7  <- chunk "lo"  (K = base + 8*half + e)
//   elems 8..15 <- chunk "hi"  (K = base + 16 + 8*half + e)
__device__ __forceinline__ v16bf frag(v8bf lo, v8bf hi) {
  v16bf r;
#pragma unroll
  for (int e = 0; e < 8; ++e) { r[e] = lo[e]; r[8 + e] = hi[e]; }
  return r;
}

// Branchless halo pixel index: candidate always computed, masked select.
__device__ __forceinline__ int halo_pix(int my, int mx, int dy, int dx, int mok) {
  int ny = my + dy;
  int nx = mx + dx;
  int cand = ny * 19 + nx;
  int ok = (int)((unsigned)ny < 19u) & (int)((unsigned)nx < 19u) & mok;
  return ok ? cand : 361;   // 361 = zero guard row
}

// ---------------------------------------------------------------------------
// Kernel 1: per-plane conv pipeline.
// grid = 4096 (= B*D planes), block = 256 threads (8 wave32).
// ---------------------------------------------------------------------------
__global__ __launch_bounds__(256)
void conv_pipeline_kernel(const float* __restrict__ xg,
    const float* __restrict__ w1g, const float* __restrict__ b1g,
    const float* __restrict__ g1g, const float* __restrict__ be1g,
    const float* __restrict__ m1g, const float* __restrict__ v1g,
    const float* __restrict__ w2g, const float* __restrict__ b2g,
    const float* __restrict__ g2g, const float* __restrict__ be2g,
    const float* __restrict__ m2g, const float* __restrict__ v2g,
    const float* __restrict__ w3g, const float* __restrict__ b3g,
    const float* __restrict__ g3g, const float* __restrict__ be3g,
    const float* __restrict__ m3g, const float* __restrict__ v3g,
    const float* __restrict__ w4g, const float* __restrict__ b4g,
    const float* __restrict__ sawg, const float* __restrict__ sabg,
    __bf16* __restrict__ hb)
{
  __shared__ float  xpad[441];          // 21x21 zero-padded input plane
  __shared__ __bf16 h1[362 * 32];       // conv1 out (row 361 = zero guard); reused as h3
  __shared__ __bf16 h2[362 * 64];       // conv2 out (row 361 = zero guard)
  __shared__ __bf16 wsh[9 * 64 * 32];   // staged weights: conv2 [t][64][32] -> conv3 [t][32][64]
  __shared__ float  w1s[288];
  __shared__ float  w4s[32];
  __shared__ float  sc1[32], sh1[32], sc2[64], sh2[64], sc3[32], sh3[32];
  __shared__ float  sbuf[361];
  __shared__ float  red[256];
  __shared__ float  attv;

  const int tid = threadIdx.x;
  const int blk = blockIdx.x;
  const int bb  = blk >> 8;    // batch index
  const int dd  = blk & 255;   // feature-plane index

  // ---- init: zero pad ring + guard rows, fold BN params, stage conv1/conv4/conv2 weights
  for (int i = tid; i < 441; i += 256) xpad[i] = 0.0f;
  if (tid < 32) h1[361 * 32 + tid] = (__bf16)0.0f;
  if (tid < 64) h2[361 * 64 + tid] = (__bf16)0.0f;
  if (tid < 288) w1s[tid] = w1g[tid];
  if (tid < 32)  w4s[tid] = w4g[tid];
  if (tid < 32) {
    float s1 = g1g[tid] * rsqrtf(v1g[tid] + EPSV);
    sc1[tid] = s1; sh1[tid] = (b1g[tid] - m1g[tid]) * s1 + be1g[tid];
    float s3 = g3g[tid] * rsqrtf(v3g[tid] + EPSV);
    sc3[tid] = s3; sh3[tid] = (b3g[tid] - m3g[tid]) * s3 + be3g[tid];
  }
  if (tid < 64) {
    float s2 = g2g[tid] * rsqrtf(v2g[tid] + EPSV);
    sc2[tid] = s2; sh2[tid] = (b2g[tid] - m2g[tid]) * s2 + be2g[tid];
  }
  // conv2 weights -> LDS as bf16 [tap][out64][in32]
  for (int j = tid; j < 9 * 64 * 32; j += 256) {
    int t = j >> 11;
    int r = j & 2047;
    int o = r >> 5;
    int i = r & 31;
    wsh[j] = (__bf16)w2g[(o * 32 + i) * 9 + t];
  }
  __syncthreads();

  // ---- gather this plane (stride-D global loads) into padded LDS tile
  for (int p = tid; p < 361; p += 256) {
    int y = p / 19, xq = p - y * 19;
    xpad[(y + 1) * 21 + (xq + 1)] = xg[((size_t)(bb * 361 + p)) * 256 + dd];
  }
  __syncthreads();

  // ---- conv1 (1->32, K=9 so plain VALU) + folded BN + exact GELU -> h1 (bf16)
  for (int idx = tid; idx < 32 * 361; idx += 256) {
    int c = idx / 361;
    int p = idx - c * 361;
    int y = p / 19, xq = p - y * 19;
    const float* wp = &w1s[c * 9];
    float acc = 0.0f;
#pragma unroll
    for (int t = 0; t < 9; ++t)
      acc += wp[t] * xpad[(y + t / 3) * 21 + (xq + t % 3)];
    h1[p * 32 + c] = (__bf16)gelu_f(acc * sc1[c] + sh1[c]);
  }
  __syncthreads();

  const int wv   = tid >> 5;
  const int lane = tid & 31;
  const int half = lane >> 4;
  const int lr   = lane & 15;

  // ---- conv2 (32->64): implicit GEMM M=361 N=64 K=9*32.
  // Each wave owns one 16-row M tile x all 4 N tiles (4 accumulators).
  // A fragments are software-pipelined across taps so the ds_load latency
  // hides behind the previous tap's 4 WMMAs (B frags live in registers).
  for (int mt = wv; mt < 23; mt += 8) {
    int m0 = mt * 16;
    int m  = m0 + lr;
    int my = m / 19, mx = m - my * 19;
    int mok = (int)(m < 361);
    v8f acc[4];
#pragma unroll
    for (int nt = 0; nt < 4; ++nt)
      acc[nt] = (v8f){0.f, 0.f, 0.f, 0.f, 0.f, 0.f, 0.f, 0.f};

    v16bf a_cur, a_nxt;
    {
      int pix = halo_pix(my, mx, -1, -1, mok);
      const v8bf* ar = (const v8bf*)&h1[pix * 32];
      a_cur = frag(ar[half], ar[2 + half]);
    }
#pragma unroll
    for (int t = 0; t < 9; ++t) {
      if (t < 8) {   // prefetch next tap's A fragment
        int pix = halo_pix(my, mx, (t + 1) / 3 - 1, (t + 1) % 3 - 1, mok);
        const v8bf* ar = (const v8bf*)&h1[pix * 32];
        a_nxt = frag(ar[half], ar[2 + half]);
      }
#pragma unroll
      for (int nt = 0; nt < 4; ++nt) {
        const v8bf* br = (const v8bf*)&wsh[(t * 64 + nt * 16 + lr) * 32];
        v16bf b = frag(br[half], br[2 + half]);
        acc[nt] = __builtin_amdgcn_wmma_f32_16x16x32_bf16(false, a_cur, false, b,
                                                          (short)0, acc[nt], false, false);
      }
      a_cur = a_nxt;
    }
#pragma unroll
    for (int nt = 0; nt < 4; ++nt) {
      int n = nt * 16 + lr;
      float sc = sc2[n], sh = sh2[n];
#pragma unroll
      for (int r = 0; r < 8; ++r) {
        int mm = m0 + r + 8 * half;
        if (mm < 361)
          h2[mm * 64 + n] = (__bf16)gelu_f(acc[nt][r] * sc + sh);
      }
    }
  }
  __syncthreads();

  // restage weights for conv3 -> LDS as bf16 [tap][out32][in64]
  for (int j = tid; j < 9 * 32 * 64; j += 256) {
    int t = j >> 11;
    int r = j & 2047;
    int o = r >> 6;
    int i = r & 63;
    wsh[j] = (__bf16)w3g[(o * 64 + i) * 9 + t];
  }
  __syncthreads();

  // ---- conv3 (64->32): M=361 N=32 K=9*64; h3 overlays the h1 buffer.
  // Each wave: one M tile x both N tiles; both channel-group A fragments
  // pipelined one tap ahead (4 WMMAs per tap to hide the loads behind).
  for (int mt = wv; mt < 23; mt += 8) {
    int m0 = mt * 16;
    int m  = m0 + lr;
    int my = m / 19, mx = m - my * 19;
    int mok = (int)(m < 361);
    v8f acc[2];
#pragma unroll
    for (int nt = 0; nt < 2; ++nt)
      acc[nt] = (v8f){0.f, 0.f, 0.f, 0.f, 0.f, 0.f, 0.f, 0.f};

    v16bf a0_cur, a1_cur, a0_nxt, a1_nxt;
    {
      int pix = halo_pix(my, mx, -1, -1, mok);
      const v8bf* ar = (const v8bf*)&h2[pix * 64];
      a0_cur = frag(ar[half],     ar[2 + half]);
      a1_cur = frag(ar[4 + half], ar[6 + half]);
    }
#pragma unroll
    for (int t = 0; t < 9; ++t) {
      if (t < 8) {   // prefetch next tap's two A fragments
        int pix = halo_pix(my, mx, (t + 1) / 3 - 1, (t + 1) % 3 - 1, mok);
        const v8bf* ar = (const v8bf*)&h2[pix * 64];
        a0_nxt = frag(ar[half],     ar[2 + half]);
        a1_nxt = frag(ar[4 + half], ar[6 + half]);
      }
#pragma unroll
      for (int nt = 0; nt < 2; ++nt) {
        const v8bf* br = (const v8bf*)&wsh[(t * 32 + nt * 16 + lr) * 64];
        v16bf b0 = frag(br[half],     br[2 + half]);
        v16bf b1 = frag(br[4 + half], br[6 + half]);
        acc[nt] = __builtin_amdgcn_wmma_f32_16x16x32_bf16(false, a0_cur, false, b0,
                                                          (short)0, acc[nt], false, false);
        acc[nt] = __builtin_amdgcn_wmma_f32_16x16x32_bf16(false, a1_cur, false, b1,
                                                          (short)0, acc[nt], false, false);
      }
      a0_cur = a0_nxt;
      a1_cur = a1_nxt;
    }
#pragma unroll
    for (int nt = 0; nt < 2; ++nt) {
      int n = nt * 16 + lr;
      float sc = sc3[n], sh = sh3[n];
#pragma unroll
      for (int r = 0; r < 8; ++r) {
        int mm = m0 + r + 8 * half;
        if (mm < 361)
          h1[mm * 32 + n] = (__bf16)gelu_f(acc[nt][r] * sc + sh);  // h3
      }
    }
  }
  __syncthreads();

  // ---- 1x1 head conv (32->1) + GELU, global-average sigmoid gate
  float b4v = b4g[0];
  float lsum = 0.0f;
  for (int p = tid; p < 361; p += 256) {
    const __bf16* hr = &h1[p * 32];
    float acc = b4v;
#pragma unroll
    for (int c = 0; c < 32; ++c) acc += w4s[c] * (float)hr[c];
    float gq = gelu_f(acc);
    sbuf[p] = gq;
    lsum += gq;
  }
  red[tid] = lsum;
  __syncthreads();
#pragma unroll
  for (int s = 128; s > 0; s >>= 1) {
    if (tid < s) red[tid] += red[tid + s];
    __syncthreads();
  }
  if (tid == 0) {
    float z = red[0] * (1.0f / 361.0f) * sawg[0] + sabg[0];
    attv = 1.0f / (1.0f + expf(-z));
  }
  __syncthreads();
  float att = attv;
  // unfold: plane scalar -> column dd of the (B*P, D) activation matrix (bf16)
  for (int p = tid; p < 361; p += 256)
    hb[((size_t)(bb * 361 + p)) * 256 + dd] = (__bf16)(sbuf[p] * att);
}

// ---------------------------------------------------------------------------
// Kernel 2: out = h @ wo^T + bo + x   (M=5776, N=256, K=256)
// grid = (361, 2): blockIdx.x = 16-row M tile, blockIdx.y = 128-col N block.
// All 8 A fragments (256B per lane: half of a 512B bf16 row) are loaded from global
// BEFORE the barrier so their latency overlaps the cooperative wo staging;
// the K loop is then pure LDS-B + WMMA.
// ---------------------------------------------------------------------------
__global__ __launch_bounds__(256)
void proj_kernel(const __bf16* __restrict__ hbuf, const float* __restrict__ wo,
                 const float* __restrict__ bo, const float* __restrict__ xg,
                 float* __restrict__ outg)
{
  __shared__ __bf16 wos[128 * 256];  // 64 KB: this block's 128 output columns
  const int tid = threadIdx.x;
  const int m0  = blockIdx.x * 16;
  const int nb  = blockIdx.y;

  const int wv = tid >> 5, lane = tid & 31, half = lane >> 4, lr = lane & 15;
  const int m  = m0 + lr;                               // always < 5776 (361*16)

  // Issue all A-fragment global loads up front (overlap with LDS staging).
  const v8bf* arow = (const v8bf*)(hbuf + (size_t)m * 256);
  v16bf afr[8];
#pragma unroll
  for (int kb = 0; kb < 8; ++kb)
    afr[kb] = frag(arow[kb * 4 + half], arow[kb * 4 + 2 + half]);

  // Cooperative staging of this block's wo slice: B[k][n] = wo[n][k] (bf16).
  for (int i = tid; i < 128 * 256; i += 256) {
    int nl = i >> 8;
    int k  = i & 255;
    wos[i] = (__bf16)wo[((size_t)(nb * 128 + nl)) * 256 + k];
  }
  __syncthreads();

  const v8bf* brow = (const v8bf*)&wos[(wv * 16 + lr) * 256];
  v8f acc = {0.f, 0.f, 0.f, 0.f, 0.f, 0.f, 0.f, 0.f};
#pragma unroll
  for (int kb = 0; kb < 8; ++kb) {
    v16bf b = frag(brow[kb * 4 + half], brow[kb * 4 + 2 + half]);
    acc = __builtin_amdgcn_wmma_f32_16x16x32_bf16(false, afr[kb], false, b,
                                                  (short)0, acc, false, false);
  }
  const int n = nb * 128 + wv * 16 + lr;
  float bov = bo[n];
#pragma unroll
  for (int r = 0; r < 8; ++r) {
    int mm = m0 + r + 8 * half;
    size_t off = (size_t)mm * 256 + n;
    outg[off] = acc[r] + bov + xg[off];   // + bias + fp32 residual
  }
}

// ---------------------------------------------------------------------------
extern "C" void kernel_launch(void* const* d_in, const int* in_sizes, int n_in,
                              void* d_out, int out_size, void* d_ws, size_t ws_size,
                              hipStream_t stream) {
  const float* x   = (const float*)d_in[0];
  const float* w1  = (const float*)d_in[1];
  const float* b1  = (const float*)d_in[2];
  const float* g1  = (const float*)d_in[3];
  const float* be1 = (const float*)d_in[4];
  const float* m1  = (const float*)d_in[5];
  const float* v1  = (const float*)d_in[6];
  const float* w2  = (const float*)d_in[7];
  const float* b2  = (const float*)d_in[8];
  const float* g2  = (const float*)d_in[9];
  const float* be2 = (const float*)d_in[10];
  const float* m2  = (const float*)d_in[11];
  const float* v2  = (const float*)d_in[12];
  const float* w3  = (const float*)d_in[13];
  const float* b3  = (const float*)d_in[14];
  const float* g3  = (const float*)d_in[15];
  const float* be3 = (const float*)d_in[16];
  const float* m3  = (const float*)d_in[17];
  const float* v3  = (const float*)d_in[18];
  const float* w4  = (const float*)d_in[19];
  const float* b4  = (const float*)d_in[20];
  const float* saw = (const float*)d_in[21];
  const float* sab = (const float*)d_in[22];
  const float* wo  = (const float*)d_in[23];
  const float* bo  = (const float*)d_in[24];

  __bf16* hb = (__bf16*)d_ws;  // (B*P, D) bf16 activation matrix: 5776*256*2 B

  conv_pipeline_kernel<<<dim3(4096), dim3(256), 0, stream>>>(
      x, w1, b1, g1, be1, m1, v1,
      w2, b2, g2, be2, m2, v2,
      w3, b3, g3, be3, m3, v3,
      w4, b4, saw, sab, hb);

  proj_kernel<<<dim3(361, 2), dim3(256), 0, stream>>>(hb, wo, bo, x, (float*)d_out);
}
